// LocalMHA_25752623907466
// MI455X (gfx1250) — matrix operand order repaired and checked by reading, not compiled, not run on hardware
//
#include <hip/hip_runtime.h>
#include <hip/hip_bf16.h>

// ---------------------------------------------------------------------------
// LocalMHA for MI455X (gfx1250).  GEMMs on v_wmma_f32_16x16x32_f16; weight /
// K / V tiles staged into LDS with GLOBAL_LOAD_ASYNC_TO_LDS_B128 (ASYNCcnt),
// double-buffered so the matrix pipe never waits on L2 latency.  K-loops
// fully unrolled to avoid sunk loop-exit accumulator copies.
// ---------------------------------------------------------------------------

typedef __attribute__((ext_vector_type(16))) _Float16 v16h;
typedef __attribute__((ext_vector_type(8)))  _Float16 v8h;
typedef __attribute__((ext_vector_type(8)))  float    v8f;
typedef __attribute__((ext_vector_type(4)))  int      v4i;

#define DIMC   1024
#define T_LEN  8192
#define BATCH  4
#define HEADS  16
#define DH     64
#define WIN    128
#define NWIN   (T_LEN / WIN)     // 64
#define BT     (BATCH * T_LEN)   // 32768

// ---- gfx1250 async copy to LDS ---------------------------------------------
#if defined(__has_builtin)
#if __has_builtin(__builtin_amdgcn_global_load_async_to_lds_b128)
#define HAVE_ASYNC_LDS 1
#endif
#if __has_builtin(__builtin_amdgcn_s_wait_asynccnt)
#define HAVE_WAIT_ASYNC 1
#endif
#endif

typedef __attribute__((address_space(1))) v4i g_v4i;
typedef __attribute__((address_space(3))) v4i l_v4i;

__device__ __forceinline__ void async_copy16(const void* g, void* l) {
#if defined(HAVE_ASYNC_LDS)
  __builtin_amdgcn_global_load_async_to_lds_b128((g_v4i*)g, (l_v4i*)l, 0, 0);
#else
  *(v8h*)l = *(const v8h*)g;   // sync fallback
#endif
}

#if defined(HAVE_WAIT_ASYNC)
#define WAIT_ASYNC(n) __builtin_amdgcn_s_wait_asynccnt(n)
#else
#define WAIT_ASYNC(n) asm volatile("s_wait_asynccnt %0" ::"n"(n) : "memory")
#endif

__device__ __forceinline__ int lane_id() { return (int)(threadIdx.x & 31u); }

// A-matrix fragment, 16x32 f16, row-major source, leading dim `ld` (halves).
// lanes 0-15: row=lane, K={0..7,16..23}; lanes 16-31: row=lane-16, K={8..15,24..31}
__device__ __forceinline__ v16h load_frag_a(const _Float16* p0, int ld) {
  int l = lane_id();
  const _Float16* p = p0 + (l & 15) * ld + ((l >> 4) << 3);
  v8h lo = *(const v8h*)(p);
  v8h hi = *(const v8h*)(p + 16);
  v16h f;
#pragma unroll
  for (int i = 0; i < 8; ++i) { f[i] = lo[i]; f[i + 8] = hi[i]; }
  return f;
}

// B-matrix fragment, 32x16 f16, source is N x K row-major.
// lanes 0-15: n=lane, K={0..7,16..23}; lanes 16-31: n=lane-16, K={8..15,24..31}
__device__ __forceinline__ v16h load_frag_b(const _Float16* p0, int ld) {
  int l = lane_id();
  const _Float16* p = p0 + (l & 15) * ld + ((l >> 4) << 3);
  v8h lo = *(const v8h*)(p);
  v8h hi = *(const v8h*)(p + 16);
  v16h f;
#pragma unroll
  for (int i = 0; i < 8; ++i) { f[i] = lo[i]; f[i + 8] = hi[i]; }
  return f;
}

__device__ __forceinline__ v8f wmma16(v16h a, v16h b, v8f c) {
  return __builtin_amdgcn_wmma_f32_16x16x32_f16(false, a, false, b, (short)0, c,
                                                false, false);
}

// ---------------------------------------------------------------------------
// fp32 -> f16 weight conversion
// ---------------------------------------------------------------------------
__global__ void cvt_f16_kernel(const float* __restrict__ a,
                               _Float16* __restrict__ o, int n) {
  int i = blockIdx.x * 256 + threadIdx.x;
  if (i < n) o[i] = (_Float16)a[i];
}

// ---------------------------------------------------------------------------
// LayerNorm over channels.  x:(B,C,T) f32 -> h:(B*T, C) f16
// ---------------------------------------------------------------------------
__global__ void ln_kernel(const float* __restrict__ x,
                          const float* __restrict__ ln_w,
                          const float* __restrict__ ln_b,
                          _Float16* __restrict__ h) {
  int tok0 = blockIdx.x * 32;               // over B*T
  int b    = tok0 >> 13;
  int t0   = tok0 & (T_LEN - 1);
  int tx = threadIdx.x, ty = threadIdx.y;
  int t  = t0 + tx;
  const float* xb = x + (size_t)b * DIMC * T_LEN;

  float s = 0.f, s2 = 0.f;
  for (int c = ty; c < DIMC; c += 8) {
    float v = xb[(size_t)c * T_LEN + t];
    s += v; s2 += v * v;
  }
  __shared__ float red[2][8][32];
  __shared__ float mu_s[32], inv_s[32];
  red[0][ty][tx] = s;
  red[1][ty][tx] = s2;
  __syncthreads();
  if (ty == 0) {
    float a = 0.f, q = 0.f;
#pragma unroll
    for (int i = 0; i < 8; ++i) { a += red[0][i][tx]; q += red[1][i][tx]; }
    float mu  = a * (1.0f / DIMC);
    float var = q * (1.0f / DIMC) - mu * mu;
    mu_s[tx]  = mu;
    inv_s[tx] = rsqrtf(var + 1e-5f);
  }
  __syncthreads();

  __shared__ _Float16 tile[32][33];
  float mu = mu_s[tx], inv = inv_s[tx];
  for (int c0 = 0; c0 < DIMC; c0 += 32) {
#pragma unroll
    for (int i = 0; i < 4; ++i) {
      int c = c0 + ty * 4 + i;
      float v = xb[(size_t)c * T_LEN + t];
      tile[c - c0][tx] = (_Float16)((v - mu) * inv * ln_w[c] + ln_b[c]);
    }
    __syncthreads();
#pragma unroll
    for (int i = 0; i < 4; ++i) {
      int tt = ty * 4 + i;
      h[(size_t)(tok0 + tt) * DIMC + c0 + tx] = tile[tx][tt];
    }
    __syncthreads();
  }
}

// ---------------------------------------------------------------------------
// QKV GEMM: qkv = h(BT x 1024) @ w_qkv^T(1024 x 3072).  Weight tile (64N x
// 32K) staged in LDS via async copies, double-buffered; A fragments software-
// pipelined one K-step ahead; K-loop fully unrolled.  Epilogue: RoPE, v^T.
// ---------------------------------------------------------------------------
__global__ void __launch_bounds__(256)
qkv_gemm_kernel(const _Float16* __restrict__ h,
                const _Float16* __restrict__ wq,
                _Float16* __restrict__ qout,
                _Float16* __restrict__ kout,
                _Float16* __restrict__ vout) {
  __shared__ __align__(16) _Float16 Bs[2][64][32];
  int tid   = (int)threadIdx.x;
  int wave  = tid >> 5;
  int mWave = blockIdx.x * 128 + wave * 16;
  int nBase = blockIdx.y * 64;

  // each thread owns one 16B chunk of the 64x32 B tile
  int brow = tid >> 2, bch = (tid & 3) * 8;
  const _Float16* gB = wq + (size_t)(nBase + brow) * DIMC + bch;
  const _Float16* hRow = h + (size_t)mWave * DIMC;

  async_copy16(gB, &Bs[0][brow][bch]);   // fill buffer 0

  v16h a = load_frag_a(hRow, DIMC);
  v8f acc[4] = {};

#pragma unroll
  for (int kb = 0; kb < 32; ++kb) {
    int cur = kb & 1;                    // compile-time after unroll
    __syncthreads();                     // readers done with fill buffer
    if (kb + 1 < 32)
      async_copy16(gB + (kb + 1) * 32, &Bs[cur ^ 1][brow][bch]);
    if (kb + 1 < 32) { WAIT_ASYNC(1); } else { WAIT_ASYNC(0); }
    __syncthreads();                     // compute buffer ready
    v16h an = a;
    if (kb + 1 < 32)
      an = load_frag_a(hRow + (kb + 1) * 32, DIMC);
#pragma unroll
    for (int j = 0; j < 4; ++j) {
      v16h bf = load_frag_b(&Bs[cur][j * 16][0], 32);
      acc[j] = wmma16(a, bf, acc[j]);
    }
    a = an;
  }

  int l = lane_id();
  int half = l >> 4, nloc = l & 15;
  int seg  = nBase >> 10;              // 0=q 1=k 2=v
  int head = (nBase & 1023) >> 6;

  // RoPE inverse frequencies depend only on lane, hoist out of row loop
  float invf0 = __expf((float)nloc * -0.28782313662425572f);
  float invf1 = __expf((float)(16 + nloc) * -0.28782313662425572f);

#pragma unroll
  for (int r = 0; r < 8; ++r) {
    int tok = mWave + r + half * 8;
    int b   = tok >> 13;
    int tt  = tok & (T_LEN - 1);
    int win = tt >> 7;
    int pos = tt & (WIN - 1);
    size_t bhw = ((size_t)(b * HEADS + head) * NWIN + win) * (WIN * DH);
    if (seg == 2) {
#pragma unroll
      for (int j = 0; j < 4; ++j) {
        int d = j * 16 + nloc;
        vout[bhw + (size_t)d * WIN + pos] = (_Float16)acc[j][r];  // transposed
      }
    } else {
      _Float16* dst = (seg == 0) ? qout : kout;
      float scale   = (seg == 0) ? 0.125f : 1.0f;   // fold 1/sqrt(64) into q
#pragma unroll
      for (int j = 0; j < 2; ++j) {
        int d = j * 16 + nloc;                      // d < 32
        float ang = (float)pos * (j == 0 ? invf0 : invf1);
        float cs, sn;
        __sincosf(ang, &sn, &cs);
        float x1 = acc[j][r], x2 = acc[j + 2][r];
        dst[bhw + (size_t)pos * DH + d]      = (_Float16)((x1 * cs - x2 * sn) * scale);
        dst[bhw + (size_t)pos * DH + d + 32] = (_Float16)((x2 * cs + x1 * sn) * scale);
      }
    }
  }
}

// ---------------------------------------------------------------------------
// Windowed attention: one block per (b,head,window).  K and V^T staged in LDS
// once via async copies (shared by all 8 waves); Q direct per-wave.
// ---------------------------------------------------------------------------
__global__ void __launch_bounds__(256)
attn_kernel(const _Float16* __restrict__ q,
            const _Float16* __restrict__ k,
            const _Float16* __restrict__ vT,
            _Float16* __restrict__ ao) {
  int id   = blockIdx.x;            // (b*HEADS+head)*NWIN + win
  int bh   = id >> 6;
  int win  = id & (NWIN - 1);
  int b    = bh >> 4;
  int head = bh & 15;

  size_t base = (size_t)id * (WIN * DH);
  const _Float16* Q = q  + base;
  const _Float16* K = k  + base;    // 128 x 64
  const _Float16* V = vT + base;    // 64 x 128 (d-major)

  __shared__ __align__(16) _Float16 Ks[WIN][DH];     // 16 KB
  __shared__ __align__(16) _Float16 Vs[DH][WIN];     // 16 KB
  __shared__ __align__(16) _Float16 As[8][16][WIN];  // 32 KB

  int tid = (int)threadIdx.x;
#pragma unroll
  for (int i = 0; i < 4; ++i) {                      // K: 1024 x 16B chunks
    int c = tid + i * 256;
    int row = c >> 3, ch = (c & 7) * 8;
    async_copy16(K + (size_t)row * DH + ch, &Ks[row][ch]);
  }
#pragma unroll
  for (int i = 0; i < 4; ++i) {                      // V: 1024 x 16B chunks
    int c = tid + i * 256;
    int row = c >> 4, ch = (c & 15) * 8;
    async_copy16(V + (size_t)row * WIN + ch, &Vs[row][ch]);
  }
  WAIT_ASYNC(0);
  __syncthreads();

  int wave  = tid >> 5;
  int l     = lane_id();
  int half  = l >> 4, nloc = l & 15;
  int mBase = wave * 16;

  // ---- scores: 16x128 strip per wave ----
  v8f sc[8] = {};
#pragma unroll
  for (int k0 = 0; k0 < DH; k0 += 32) {
    v16h a = load_frag_a(Q + (size_t)mBase * DH + k0, DH);
#pragma unroll
    for (int j = 0; j < 8; ++j) {
      v16h bf = load_frag_b(&Ks[j * 16][k0], DH);
      sc[j] = wmma16(a, bf, sc[j]);
    }
  }

  // ---- softmax rows + stage attn (f16) in LDS ----
#pragma unroll
  for (int r = 0; r < 8; ++r) {
    float m = -3.0e38f;
#pragma unroll
    for (int j = 0; j < 8; ++j) m = fmaxf(m, sc[j][r]);
    for (int off = 1; off < 16; off <<= 1) m = fmaxf(m, __shfl_xor(m, off, 32));
    float p[8], sum = 0.f;
#pragma unroll
    for (int j = 0; j < 8; ++j) { p[j] = __expf(sc[j][r] - m); sum += p[j]; }
    for (int off = 1; off < 16; off <<= 1) sum += __shfl_xor(sum, off, 32);
    float rinv = 1.0f / sum;
    int row = r + half * 8;
#pragma unroll
    for (int j = 0; j < 8; ++j)
      As[wave][row][j * 16 + nloc] = (_Float16)(p[j] * rinv);
  }
  __syncthreads();

  // ---- out = attn(16x128) @ V^T ----
  v8f oa[4] = {};
  int ar = l & 15, ao8 = (l >> 4) << 3;
#pragma unroll
  for (int k0 = 0; k0 < WIN; k0 += 32) {
    v16h a;
    {
      v8h lo = *(const v8h*)(&As[wave][ar][k0 + ao8]);
      v8h hi = *(const v8h*)(&As[wave][ar][k0 + ao8 + 16]);
#pragma unroll
      for (int i = 0; i < 8; ++i) { a[i] = lo[i]; a[i + 8] = hi[i]; }
    }
#pragma unroll
    for (int dt = 0; dt < 4; ++dt) {
      v16h bf = load_frag_b(&Vs[dt * 16][k0], WIN);
      oa[dt] = wmma16(a, bf, oa[dt]);
    }
  }

  // ---- store to ao (B*T, C) f16 ----
#pragma unroll
  for (int r = 0; r < 8; ++r) {
    int m = mBase + r + half * 8;
    size_t rowoff = ((size_t)(b * T_LEN + win * WIN + m)) * DIMC + head * DH;
#pragma unroll
    for (int dt = 0; dt < 4; ++dt)
      ao[rowoff + dt * 16 + nloc] = (_Float16)oa[dt][r];
  }
}

// ---------------------------------------------------------------------------
// Output GEMM: out = ao(BT x 1024) @ w_out^T(1024 x 1024) + residual,
// written (B, C, T) f32 via LDS-transposed, coalesced epilogue.
// ---------------------------------------------------------------------------
__global__ void __launch_bounds__(256)
out_proj_kernel(const _Float16* __restrict__ ao,
                const _Float16* __restrict__ wo,
                const float* __restrict__ x,
                float* __restrict__ out) {
  __shared__ __align__(16) _Float16 Bs[2][64][32];
  int tid   = (int)threadIdx.x;
  int wave  = tid >> 5;
  int mWave = blockIdx.x * 128 + wave * 16;
  int nBase = blockIdx.y * 64;

  int brow = tid >> 2, bch = (tid & 3) * 8;
  const _Float16* gB = wo + (size_t)(nBase + brow) * DIMC + bch;
  const _Float16* aRow = ao + (size_t)mWave * DIMC;

  async_copy16(gB, &Bs[0][brow][bch]);

  v16h a = load_frag_a(aRow, DIMC);
  v8f acc[4] = {};

#pragma unroll
  for (int kb = 0; kb < 32; ++kb) {
    int cur = kb & 1;
    __syncthreads();
    if (kb + 1 < 32)
      async_copy16(gB + (kb + 1) * 32, &Bs[cur ^ 1][brow][bch]);
    if (kb + 1 < 32) { WAIT_ASYNC(1); } else { WAIT_ASYNC(0); }
    __syncthreads();
    v16h an = a;
    if (kb + 1 < 32)
      an = load_frag_a(aRow + (kb + 1) * 32, DIMC);
#pragma unroll
    for (int j = 0; j < 4; ++j) {
      v16h bf = load_frag_b(&Bs[cur][j * 16][0], 32);
      acc[j] = wmma16(a, bf, acc[j]);
    }
    a = an;
  }

  // stage block tile (64 c x 128 t) in LDS, then coalesced stores along T
  __shared__ float otile[64][129];
  int l = lane_id();
  int half = l >> 4, nloc = l & 15;
#pragma unroll
  for (int r = 0; r < 8; ++r) {
    int tLoc = wave * 16 + r + half * 8;
#pragma unroll
    for (int j = 0; j < 4; ++j)
      otile[j * 16 + nloc][tLoc] = acc[j][r];
  }
  __syncthreads();

  int tok0  = blockIdx.x * 128;
  int b     = tok0 >> 13;
  int tBase = tok0 & (T_LEN - 1);
  for (int e = tid; e < 64 * 128; e += 256) {
    int cr = e >> 7, tc = e & 127;
    size_t idx = ((size_t)b * DIMC + nBase + cr) * T_LEN + tBase + tc;
    out[idx] = otile[cr][tc] + x[idx];
  }
}

// ---------------------------------------------------------------------------
// Launch
// ---------------------------------------------------------------------------
extern "C" void kernel_launch(void* const* d_in, const int* in_sizes, int n_in,
                              void* d_out, int out_size, void* d_ws, size_t ws_size,
                              hipStream_t stream) {
  const float* x     = (const float*)d_in[0];
  const float* ln_w  = (const float*)d_in[1];
  const float* ln_b  = (const float*)d_in[2];
  const float* w_qkv = (const float*)d_in[3];
  const float* w_out = (const float*)d_in[4];
  float* out = (float*)d_out;

  char* ws = (char*)d_ws;
  _Float16* h  = (_Float16*)(ws + 0);           // 64 MB  (BT x 1024 f16)
  _Float16* wq = (_Float16*)(ws + 67108864);    //  6 MB
  _Float16* wo = (_Float16*)(ws + 73400320);    //  2 MB
  _Float16* qb = (_Float16*)(ws + 75497472);    // 64 MB
  _Float16* kb = (_Float16*)(ws + 142606336);   // 64 MB
  _Float16* vb = (_Float16*)(ws + 209715200);   // 64 MB (transposed per window)
  _Float16* aob = h;                            // reuse h after qkv gemm

  cvt_f16_kernel<<<(3 * DIMC * DIMC + 255) / 256, 256, 0, stream>>>(w_qkv, wq, 3 * DIMC * DIMC);
  cvt_f16_kernel<<<(DIMC * DIMC + 255) / 256, 256, 0, stream>>>(w_out, wo, DIMC * DIMC);

  ln_kernel<<<BT / 32, dim3(32, 8), 0, stream>>>(x, ln_w, ln_b, h);

  qkv_gemm_kernel<<<dim3(BT / 128, (3 * DIMC) / 64), 256, 0, stream>>>(h, wq, qb, kb, vb);

  attn_kernel<<<BATCH * HEADS * NWIN, 256, 0, stream>>>(qb, kb, vb, aob);

  out_proj_kernel<<<dim3(BT / 128, DIMC / 64), 256, 0, stream>>>(aob, wo, x, out);
}
